// VQEmbedding_81621558493387
// MI455X (gfx1250) — matrix operand order repaired and checked by reading, not compiled, not run on hardware
//
#include <hip/hip_runtime.h>
#include <hip/hip_bf16.h>
#include <stdint.h>

// ---------------------------------------------------------------------------
// VQ-VAE quantization for MI455X (gfx1250, wave32, WMMA).
//  z:        (64, 256, 32, 32) f32   -> rows N = 65536, D = 256
//  codebook: (1024, 256) f32         -> K = 1024
// Outputs (concatenated in d_out, f32):
//  [0, 16777216)            quantized_st (B,D,H,W)  == codebook[idx] (fwd value)
//  [16777216, +65536)       indices as float
//  [+65536, +1)             loss = 1.25 * sum((q - z)^2)
// ---------------------------------------------------------------------------

typedef __bf16          bf16x16 __attribute__((ext_vector_type(16)));
typedef unsigned short  u16x16  __attribute__((ext_vector_type(16)));
typedef unsigned short  u16x8   __attribute__((ext_vector_type(8)));
typedef float           v8f     __attribute__((ext_vector_type(8)));
typedef int             v4i     __attribute__((ext_vector_type(4)));

#define NB    64
#define ND    256
#define NHW   1024           // 32*32
#define NK    1024
#define NROWS (NB * NHW)     // 65536
#define OUT_Q    0
#define OUT_IDX  (NB * ND * NHW)      // 16777216
#define OUT_LOSS (OUT_IDX + NROWS)

#define TILE_CODES 32
#define NT         (NK / TILE_CODES)     // 32 staged tiles
#define SROW       264                   // ushort stride per code row (bank-spread pad)
#define SMEM_TILE  (TILE_CODES * SROW)   // ushorts per staging buffer

// CDNA5 async-copy path (global_load_async_to_lds_b128 + s_wait_asynccnt),
// guarded so a toolchain without the builtins falls back to a register-
// pipelined staging loop (same barrier structure, latency still hidden).
// Probe result (round 2): the builtin exists and takes int4* pointers.
#if __has_builtin(__builtin_amdgcn_global_load_async_to_lds_b128) && \
    __has_builtin(__builtin_amdgcn_s_wait_asynccnt)
#define VQ_ASYNC 1
typedef __attribute__((address_space(1))) v4i* gv4i_t;   // global src
typedef __attribute__((address_space(3))) v4i* lv4i_t;   // LDS dst
#else
#define VQ_ASYNC 0
#endif

__device__ __forceinline__ unsigned short f2bf_bits(float f) {
  union { float f; unsigned u; } x; x.f = f;
  unsigned r = x.u + 0x7FFFu + ((x.u >> 16) & 1u);  // round-to-nearest-even
  return (unsigned short)(r >> 16);
}

// ---------------------------------------------------------------------------
// K0: per-code squared norm + one-time codebook f32 -> bf16 conversion.
// ---------------------------------------------------------------------------
__global__ __launch_bounds__(256) void vq_prep_kernel(
    const float* __restrict__ cb, float* __restrict__ cnorm,
    unsigned short* __restrict__ cbh)
{
  __shared__ float red[256];
  const int k = blockIdx.x, d = threadIdx.x;
  const float v = cb[k * ND + d];
  cbh[k * ND + d] = f2bf_bits(v);
  red[d] = v * v;
  __syncthreads();
  for (int s = 128; s > 0; s >>= 1) {
    if (d < s) red[d] += red[d + s];
    __syncthreads();
  }
  if (d == 0) cnorm[k] = red[0];
}

// ---------------------------------------------------------------------------
// K1: fused distance-GEMM + argmin. 8 waves/block, 16 rows per wave.
// A (z rows, bf16) resident in VGPRs for the whole K loop; 32-code codebook
// tiles staged in LDS (async DMA double-buffered when available); 16 x
// v_wmma_f32_16x16x32_bf16 per barrier pair; argmin epilogue per 16x16 tile.
// argmin(dist) == argmin(||c||^2 - 2 z.c): the ||z||^2 term is row-constant.
// ---------------------------------------------------------------------------
__global__ __launch_bounds__(256) void vq_argmin_kernel(
    const float* __restrict__ z, const unsigned short* __restrict__ cbh,
    const float* __restrict__ cnorm, float* __restrict__ out_idx)
{
#if VQ_ASYNC
  __shared__ unsigned short smem[2 * SMEM_TILE];
#else
  __shared__ unsigned short smem[SMEM_TILE];
#endif

  const int tid  = threadIdx.x;
  const int lane = tid & 31;
  const int wave = tid >> 5;
  const int l15  = lane & 15;
  const int hiA  = (lane >> 4) ? 8  : 0;   // A-fragment K offset (ISA 16-bit A table)
  const int hiB  = (lane >> 4) ? 8  : 0;   // B-fragment K offset (ISA 16-bit B table)

  const int rowbase = (blockIdx.x * 8 + wave) * 16;

  // ---- Load A fragments once: 16 rows x 256 features, NCHW-strided ----
  const int row = rowbase + l15;
  const int zlb = ((row >> 10) << 18) + (row & 1023);   // b*D*HW + hw
  bf16x16 afrag[8];
#pragma unroll
  for (int c = 0; c < 8; ++c) {
    const int kb = c * 32 + hiA;
    u16x16 au;
#pragma unroll
    for (int e = 0; e < 16; ++e) {
      const int kf = kb + (e & 7) + ((e & 8) ? 16 : 0);
      au[e] = f2bf_bits(z[zlb + (kf << 10)]);
    }
    afrag[c] = __builtin_bit_cast(bf16x16, au);
  }

  float best[8];
  int   bidx[8];
#pragma unroll
  for (int v = 0; v < 8; ++v) { best[v] = 3.4e38f; bidx[v] = 0; }

  // staging role: thread stages 32 consecutive bf16 feats of one code row
  const int code_l  = tid >> 3;         // 0..31 (code within tile)
  const int featblk = (tid & 7) * 32;   // 0..224 (feature offset, ushorts)

  // per-subtile WMMA + argmin epilogue (two 16-code subtiles per staged tile)
#define VQ_SUBTILE(SMBASE, COLBASE16)                                         \
  {                                                                           \
    v8f acc = {};                                                             \
    _Pragma("unroll")                                                         \
    for (int c = 0; c < 8; ++c) {                                             \
      const unsigned short* bp = &smem[(SMBASE) + c * 32 + hiB];              \
      u16x8 b0 = *(const u16x8*)bp;                                           \
      u16x8 b1 = *(const u16x8*)(bp + 16);                                    \
      u16x16 bu = __builtin_shufflevector(b0, b1,                             \
                    0,1,2,3,4,5,6,7,8,9,10,11,12,13,14,15);                   \
      bf16x16 bfrag = __builtin_bit_cast(bf16x16, bu);                        \
      acc = __builtin_amdgcn_wmma_f32_16x16x32_bf16(                          \
                false, afrag[c], false, bfrag, (short)0, acc, false, false);  \
    }                                                                         \
    const int   code = (COLBASE16) + l15;                                     \
    const float cn   = cnorm[code];                                           \
    _Pragma("unroll")                                                         \
    for (int v = 0; v < 8; ++v) {                                             \
      const float s = cn - 2.0f * acc[v];                                     \
      if (s < best[v]) { best[v] = s; bidx[v] = code; }                       \
    }                                                                         \
  }

#if VQ_ASYNC
  // ---- double-buffered async DMA staging (CDNA5 ASYNCcnt path) ----
  // 4 x b128 per thread per tile; ASYNCcnt decrements in issue order, so
  // waiting <= 4 guarantees the older tile's 4 ops have landed in LDS.
  auto issue_tile = [&](int t) {
    unsigned short* g =
        const_cast<unsigned short*>(cbh) + (t * TILE_CODES + code_l) * ND + featblk;
    unsigned short* l = &smem[(t & 1) * SMEM_TILE + code_l * SROW + featblk];
#pragma unroll
    for (int j = 0; j < 4; ++j)
      __builtin_amdgcn_global_load_async_to_lds_b128(
          (gv4i_t)(g + j * 8), (lv4i_t)(l + j * 8), 0, 0);
  };
  issue_tile(0);
  issue_tile(1);
  for (int ct = 0; ct < NT; ++ct) {
    if (ct < NT - 1) __builtin_amdgcn_s_wait_asynccnt(4);
    else             __builtin_amdgcn_s_wait_asynccnt(0);
    __syncthreads();                       // tile ct staged by all waves
    const int bufbase = (ct & 1) * SMEM_TILE;
    const int colbase = ct * TILE_CODES;
    VQ_SUBTILE(bufbase + l15 * SROW,              colbase);
    VQ_SUBTILE(bufbase + (16 + l15) * SROW,       colbase + 16);
    __syncthreads();                       // buffer free to overwrite
    if (ct + 2 < NT) issue_tile(ct + 2);
  }
#else
  // ---- register-pipelined staging: issue tile ct+1 loads before computing
  // tile ct; the loadcnt wait lands at the next iteration's ds_store, so the
  // L2 latency hides behind 16 WMMAs + epilogues.
  u16x8 r0, r1, r2, r3;
  {
    const u16x8* src = (const u16x8*)(cbh + code_l * ND + featblk);
    r0 = src[0]; r1 = src[1]; r2 = src[2]; r3 = src[3];
  }
  for (int ct = 0; ct < NT; ++ct) {
    const int colbase = ct * TILE_CODES;
    __syncthreads();                       // previous tile fully consumed
    {
      u16x8* dst = (u16x8*)&smem[code_l * SROW + featblk];
      dst[0] = r0; dst[1] = r1; dst[2] = r2; dst[3] = r3;
      if (ct + 1 < NT) {
        const u16x8* src =
            (const u16x8*)(cbh + ((colbase + TILE_CODES) + code_l) * ND + featblk);
        r0 = src[0]; r1 = src[1]; r2 = src[2]; r3 = src[3];
        __builtin_prefetch(
            cbh + ((colbase + 2 * TILE_CODES) + code_l) * ND + featblk, 0, 0);
      }
    }
    __syncthreads();                       // tile staged
    VQ_SUBTILE(l15 * SROW,           colbase);
    VQ_SUBTILE((16 + l15) * SROW,    colbase + 16);
  }
#endif
#undef VQ_SUBTILE

  // ---- cross-lane argmin within each 16-lane half (rows v and v+8) ----
#pragma unroll
  for (int v = 0; v < 8; ++v) {
    float bv = best[v];
    int   bi = bidx[v];
#pragma unroll
    for (int m = 8; m >= 1; m >>= 1) {
      const float ov = __shfl_xor(bv, m, 16);
      const int   oi = __shfl_xor(bi, m, 16);
      if (ov < bv || (ov == bv && oi < bi)) { bv = ov; bi = oi; }
    }
    if (l15 == 0) {
      const int r = rowbase + v + ((lane >> 4) ? 8 : 0);
      out_idx[r] = (float)bi;
    }
  }
}

// ---------------------------------------------------------------------------
// K2: gather codebook rows back to NCHW output + deterministic loss partials.
// Thread handles 4 consecutive hw elements of one (b, d) plane (float4 I/O).
// ---------------------------------------------------------------------------
__global__ __launch_bounds__(256) void vq_gather_kernel(
    const float* __restrict__ z, const float* __restrict__ cb,
    const float* __restrict__ idxf, float* __restrict__ outq,
    float* __restrict__ partial)
{
  __shared__ float red[256];
  const int t  = threadIdx.x;
  const int i  = (blockIdx.x * 256 + t) * 4;       // linear NCHW index
  const int b  = i >> 18;                          // / (256*1024)
  const int d  = (i >> 10) & 255;
  const int hw = i & 1023;
  const int r0 = b * NHW + hw;

  const float4 z4 = *(const float4*)(z + i);
  const float* zp = (const float*)&z4;
  float q[4], ss = 0.f;
#pragma unroll
  for (int j = 0; j < 4; ++j) {
    const int code = (int)idxf[r0 + j];
    q[j] = cb[code * ND + d];
    const float dv = q[j] - zp[j];
    ss += dv * dv;
  }
  *(float4*)(outq + i) = make_float4(q[0], q[1], q[2], q[3]);

  red[t] = ss;
  __syncthreads();
  for (int s = 128; s > 0; s >>= 1) {
    if (t < s) red[t] += red[t + s];
    __syncthreads();
  }
  if (t == 0) partial[blockIdx.x] = red[0];
}

// ---------------------------------------------------------------------------
// K3: deterministic reduction of 16384 partials -> loss scalar.
// loss = (0.25 + 1.0) * sum((q - z)^2) ; both loss terms share forward value.
// ---------------------------------------------------------------------------
__global__ __launch_bounds__(256) void vq_loss_kernel(
    const float* __restrict__ partial, float* __restrict__ loss)
{
  __shared__ float red[256];
  float s = 0.f;
  for (int i = threadIdx.x; i < 16384; i += 256) s += partial[i];
  red[threadIdx.x] = s;
  __syncthreads();
  for (int st = 128; st > 0; st >>= 1) {
    if (threadIdx.x < st) red[threadIdx.x] += red[threadIdx.x + st];
    __syncthreads();
  }
  if (threadIdx.x == 0) loss[0] = red[0] * 1.25f;
}

// ---------------------------------------------------------------------------
extern "C" void kernel_launch(void* const* d_in, const int* in_sizes, int n_in,
                              void* d_out, int out_size, void* d_ws, size_t ws_size,
                              hipStream_t stream)
{
  const float* z  = (const float*)d_in[0];   // (64,256,32,32) f32
  const float* cb = (const float*)d_in[1];   // (1024,256) f32
  float* out = (float*)d_out;

  // Workspace layout: cnorm (1024 f32) | partials (16384 f32) | cbh (262144 u16)
  float*          cnorm   = (float*)d_ws;
  float*          partial = cnorm + NK;
  unsigned short* cbh     = (unsigned short*)(partial + 16384);

  vq_prep_kernel  <<<NK,          256, 0, stream>>>(cb, cnorm, cbh);
  vq_argmin_kernel<<<NROWS / 128, 256, 0, stream>>>(z, cbh, cnorm, out + OUT_IDX);
  vq_gather_kernel<<<(NB * ND * NHW) / 1024, 256, 0, stream>>>(
      z, cb, out + OUT_IDX, out + OUT_Q, partial);
  vq_loss_kernel  <<<1,           256, 0, stream>>>(partial, out + OUT_LOSS);
}
